// TransformerEncoderLayerMoE_RoPE_4784593568355
// MI455X (gfx1250) — compile-verified, matrix operand order checked by reading, not run on hardware
//
#include <hip/hip_runtime.h>
#include <hip/hip_bf16.h>

#define S_   2048
#define B_   2
#define D_   512
#define H_   8
#define DH_  64
#define E_   8
#define F_   1024
#define NTOK (S_*B_)   // 4096

typedef __attribute__((ext_vector_type(16))) _Float16 v16h;
typedef __attribute__((ext_vector_type(8)))  _Float16 v8h;
typedef __attribute__((ext_vector_type(8)))  float    v8f;
typedef __attribute__((ext_vector_type(4)))  int      v4i;

#if defined(__has_builtin)
#  if __has_builtin(__builtin_amdgcn_global_load_async_to_lds_b128)
#    define HAVE_ASYNC 1
#  endif
#endif
#ifndef HAVE_ASYNC
#  define HAVE_ASYNC 0
#endif

// 16-byte global -> LDS async copy (ASYNCcnt) with synchronous fallback.
// Toolchain declares the builtin with generic `v4i*` pointer params
// (diagnostic-confirmed): (global_src, lds_dst, imm offset, imm cpol).
__device__ __forceinline__ void async_cp16(const _Float16* g, _Float16* l) {
#if HAVE_ASYNC
  __builtin_amdgcn_global_load_async_to_lds_b128((v4i*)g, (v4i*)l, 0, 0);
#else
  *(v8h*)l = *(const v8h*)g;
#endif
}
__device__ __forceinline__ void wait_async() {
#if HAVE_ASYNC
#  if defined(__has_builtin)
#    if __has_builtin(__builtin_amdgcn_s_wait_asynccnt)
  __builtin_amdgcn_s_wait_asynccnt(0);
#    else
  asm volatile("s_wait_asynccnt 0" ::: "memory");
#    endif
#  else
  asm volatile("s_wait_asynccnt 0" ::: "memory");
#  endif
#endif
}

__device__ __forceinline__ v8f wmma_f16(v16h a, v16h b, v8f c) {
  // D = A(16x32 f16) x B(32x16 f16) + C(16x16 f32)
  return __builtin_amdgcn_wmma_f32_16x16x32_f16(false, a, false, b, (short)0, c,
                                                false, false);
}

// Build a 16-half A/B fragment from a 32-element K-row (base points at K=0).
// CDNA5 16-bit layout: dwords 0-3 hold K = lh*8 .. lh*8+7, dwords 4-7 hold
// K = 16 + lh*8 .. 16 + lh*8 + 7   (lh = lane>>4).
__device__ __forceinline__ v16h ld_frag(const _Float16* base, int lh) {
  v8h lo = *(const v8h*)(base + lh*8);
  v8h hi = *(const v8h*)(base + 16 + lh*8);
  return __builtin_shufflevector(lo, hi, 0,1,2,3,4,5,6,7,8,9,10,11,12,13,14,15);
}

// ---------------------------------------------------------------- converts
__global__ void cvt_kernel(const float* __restrict__ a, _Float16* __restrict__ o, int n) {
  int i = blockIdx.x*256 + threadIdx.x;
  if (i < n) o[i] = (_Float16)a[i];
}
// batched fp32 (K x N) -> f16 transposed (N x K); `total` = mats*K*N
__global__ void cvt_t_kernel(const float* __restrict__ w, _Float16* __restrict__ wt,
                             int K, int N, int total) {
  int i = blockIdx.x*256 + threadIdx.x;
  if (i >= total) return;
  int mat = i / (K*N);
  int r   = i - mat*(K*N);
  int k = r / N, n = r % N;
  wt[(size_t)mat*K*N + (size_t)n*K + k] = (_Float16)w[i];
}
__global__ void zero_f32(float* p, int n) {
  int i = blockIdx.x*256 + threadIdx.x;
  if (i < n) p[i] = 0.f;
}
__global__ void zero_i32(int* p, int n) {
  int i = blockIdx.x*256 + threadIdx.x;
  if (i < n) p[i] = 0;
}

// ---------------------------------------------------------------- WMMA GEMM
// C(MxN) = A(MxK) @ B(KxN) [+bias], with B supplied TRANSPOSED: Bt is N x K.
// f16 in, f32 accum. Tile 64x64x32, 128 threads = 4 waves (2x2 wave grid,
// 2x2 wmma tiles each). Double-buffered LDS staged via async B128 copies.
// EPI 0: C = acc + bias            (optional f16 mirror Ch)
// EPI 1: gather A rows via rows[], ReLU, write f16 Ch only   (MoE up)
// EPI 2: scatter: atomicAdd(C[rows[m]], (acc+bias)*owt[m])   (MoE down)
#define BM 64
#define BN 64
#define BK 32
template<int EPI>
__global__ void __launch_bounds__(128) gemm_f16(
    const _Float16* __restrict__ A, const _Float16* __restrict__ Bt,
    const float* __restrict__ bias, float* __restrict__ C, _Float16* __restrict__ Ch,
    int M, int Nn, int K, int lda, int ldb, int ldc,
    const int* __restrict__ rows, const float* __restrict__ owt,
    const int* __restrict__ cntp)
{
  __shared__ alignas(64) _Float16 As[2][BM*BK];   // [m][k]
  __shared__ alignas(64) _Float16 Bs[2][BN*BK];   // [n][k]
  const int tid  = threadIdx.x;
  const int m0   = blockIdx.y * BM;
  const int n0   = blockIdx.x * BN;
  const int cnt  = cntp ? cntp[0] : M;
  if (m0 >= cnt) return;
  const int lane = tid & 31, wv = tid >> 5;
  const int lh   = lane >> 4, l15 = lane & 15;
  const int wm   = (wv >> 1) * 32, wn = (wv & 1) * 32;

  // stage one 64x32 A tile + 64x32 Bt tile into LDS buffer `buf`
  auto stage = [&](int k0, int buf) {
#pragma unroll
    for (int it = 0; it < 2; ++it) {
      int g   = tid + it*128;        // 256 chunks of 8 halfs
      int row = g >> 2;              // 0..63
      int kq  = (g & 3) * 8;         // 0,8,16,24
      int tr  = m0 + row;
      int ga  = tr;
      if (EPI == 1 && rows) ga = (tr < cnt) ? rows[tr] : 0;
      async_cp16(A  + (size_t)ga*lda        + k0 + kq, &As[buf][row*BK + kq]);
      async_cp16(Bt + (size_t)(n0+row)*ldb  + k0 + kq, &Bs[buf][row*BK + kq]);
    }
  };

  v8f zero = {0,0,0,0,0,0,0,0};
  v8f acc[2][2];
  acc[0][0] = zero; acc[0][1] = zero; acc[1][0] = zero; acc[1][1] = zero;

  stage(0, 0);
  wait_async();
  __syncthreads();

  const int nk = K / BK;
  for (int kt = 0; kt < nk; ++kt) {
    const int buf = kt & 1;
    if (kt + 1 < nk) stage((kt + 1)*BK, buf ^ 1);   // overlap next fetch
#pragma unroll
    for (int tm = 0; tm < 2; ++tm) {
      v16h a = ld_frag(&As[buf][(wm + tm*16 + l15)*BK], lh);
#pragma unroll
      for (int tn = 0; tn < 2; ++tn) {
        v16h b = ld_frag(&Bs[buf][(wn + tn*16 + l15)*BK], lh);
        acc[tm][tn] = wmma_f16(a, b, acc[tm][tn]);
      }
    }
    wait_async();
    __syncthreads();
  }

  // epilogue: D VGPR i -> M = i + 8*lh, N = l15 within each 16x16 tile
#pragma unroll
  for (int tm = 0; tm < 2; ++tm)
#pragma unroll
  for (int tn = 0; tn < 2; ++tn)
#pragma unroll
  for (int i = 0; i < 8; ++i) {
    int mrow = m0 + wm + tm*16 + lh*8 + i;
    int col  = n0 + wn + tn*16 + l15;
    float v  = acc[tm][tn][i];
    if (bias) v += bias[col];
    if (EPI == 0) {
      C[(size_t)mrow*ldc + col] = v;
      if (Ch) Ch[(size_t)mrow*ldc + col] = (_Float16)v;
    } else if (EPI == 1) {
      if (mrow < cnt) {
        v = fmaxf(v, 0.f);
        Ch[(size_t)mrow*ldc + col] = (_Float16)v;
      }
    } else {
      if (mrow < cnt) {
        int tok = rows[mrow];
        atomicAdd(&C[(size_t)tok*ldc + col], v * owt[mrow]);
      }
    }
  }
}

// ---------------------------------------------------------------- RoPE + relayout
// q,k,v fp32 [S,B,H,DH] -> qh,kh f16 [B,H,S,DH], vt f16 [B,H,DH,S]
__global__ void rope_reorder_kernel(const float* __restrict__ q, const float* __restrict__ k,
                                    const float* __restrict__ v,
                                    const float* __restrict__ cosb, const float* __restrict__ sinb,
                                    _Float16* __restrict__ qh, _Float16* __restrict__ kh,
                                    _Float16* __restrict__ vt)
{
  int i = blockIdx.x*256 + threadIdx.x;
  if (i >= NTOK*D_) return;
  int d = i & (DH_-1);
  int h = (i >> 6) & (H_-1);
  int b = (i >> 9) & (B_-1);
  int s = i >> 10;
  size_t si = (size_t)(s*B_ + b)*D_ + h*DH_ + d;
  size_t ri = (size_t)(s*B_ + b)*D_ + h*DH_ + ((d < 32) ? d + 32 : d - 32);
  float c  = cosb[s*DH_ + d], sn = sinb[s*DH_ + d];
  float qv = q[si], kv = k[si], vv = v[si];
  float qr = (d < 32) ? -q[ri] : q[ri];
  float kr = (d < 32) ? -k[ri] : k[ri];
  size_t bhsd = ((size_t)(b*H_ + h)*S_ + s)*DH_ + d;
  qh[bhsd] = (_Float16)(qv*c + qr*sn);
  kh[bhsd] = (_Float16)(kv*c + kr*sn);
  vt[((size_t)(b*H_ + h)*DH_ + d)*S_ + s] = (_Float16)vv;
}

// ---------------------------------------------------------------- flash attention
// One wave = 16 query rows; block = 8 waves = 128 rows. grid(S/128, B*H).
__global__ void __launch_bounds__(256) attn_kernel(
    const _Float16* __restrict__ qh, const _Float16* __restrict__ kh,
    const _Float16* __restrict__ vt, _Float16* __restrict__ ctxh)
{
  __shared__ alignas(64) _Float16 Ps[8][16*32];  // per-wave P transpose slab
  const int bh = blockIdx.y;
  const int b  = bh >> 3, h = bh & 7;
  const int wv = threadIdx.x >> 5, lane = threadIdx.x & 31;
  const int lh = lane >> 4, l15 = lane & 15;
  const int q0 = blockIdx.x*128 + wv*16;
  const _Float16* Q  = qh + (size_t)bh * S_ * DH_;
  const _Float16* Kp = kh + (size_t)bh * S_ * DH_;
  const _Float16* Vp = vt + (size_t)bh * DH_ * S_;

  v16h qa[2];
#pragma unroll
  for (int c = 0; c < 2; ++c)
    qa[c] = ld_frag(Q + (size_t)(q0 + l15)*DH_ + c*32, lh);

  v8f zero = {0,0,0,0,0,0,0,0};
  v8f o[4];
#pragma unroll
  for (int t = 0; t < 4; ++t) o[t] = zero;
  float rmax[8], rsum[8];
#pragma unroll
  for (int i = 0; i < 8; ++i) { rmax[i] = -1e30f; rsum[i] = 0.f; }
  _Float16* myP = &Ps[wv][0];

  for (int kv = 0; kv < S_; kv += 32) {
    if (kv + 32 < S_) __builtin_prefetch(Kp + (size_t)(kv + 32 + l15)*DH_, 0, 1);
    v8f s0 = zero, s1 = zero;           // score tiles: cols [kv,kv+16) / [kv+16,kv+32)
#pragma unroll
    for (int c = 0; c < 2; ++c) {
      v16h kb0 = ld_frag(Kp + (size_t)(kv      + l15)*DH_ + c*32, lh);
      v16h kb1 = ld_frag(Kp + (size_t)(kv + 16 + l15)*DH_ + c*32, lh);
      s0 = wmma_f16(qa[c], kb0, s0);
      s1 = wmma_f16(qa[c], kb1, s1);
    }
    float scl[8];
#pragma unroll
    for (int i = 0; i < 8; ++i) {
      float sa = s0[i] * 0.125f;        // 1/sqrt(DH)
      float sb = s1[i] * 0.125f;
      float m = fmaxf(sa, sb);
#pragma unroll
      for (int msk = 1; msk < 16; msk <<= 1) m = fmaxf(m, __shfl_xor(m, msk, 32));
      float nmax = fmaxf(rmax[i], m);
      scl[i] = __expf(rmax[i] - nmax);
      float p0 = __expf(sa - nmax);
      float p1 = __expf(sb - nmax);
      myP[(i + lh*8)*32      + l15] = (_Float16)p0;
      myP[(i + lh*8)*32 + 16 + l15] = (_Float16)p1;
      float ps = p0 + p1;
#pragma unroll
      for (int msk = 1; msk < 16; msk <<= 1) ps += __shfl_xor(ps, msk, 32);
      rsum[i] = rsum[i]*scl[i] + ps;
      rmax[i] = nmax;
    }
#pragma unroll
    for (int t = 0; t < 4; ++t)
#pragma unroll
      for (int i = 0; i < 8; ++i) o[t][i] *= scl[i];
    // wave-private LDS RAW: drain DS before reading P as an A-fragment
    asm volatile("s_wait_dscnt 0" ::: "memory");
    v16h pa = ld_frag(myP + l15*32, lh);
#pragma unroll
    for (int t = 0; t < 4; ++t) {
      v16h vb = ld_frag(Vp + (size_t)(t*16 + l15)*S_ + kv, lh);
      o[t] = wmma_f16(pa, vb, o[t]);
    }
  }
#pragma unroll
  for (int t = 0; t < 4; ++t)
#pragma unroll
  for (int i = 0; i < 8; ++i) {
    int srow = q0 + i + lh*8;
    float val = o[t][i] / rsum[i];
    ctxh[(size_t)(srow*B_ + b)*D_ + h*DH_ + t*16 + l15] = (_Float16)val;
  }
}

// ---------------------------------------------------------------- residual + LN
__global__ void __launch_bounds__(256) add_ln_kernel(
    const float* __restrict__ a, const float* __restrict__ b,
    const float* __restrict__ g, const float* __restrict__ be,
    float* __restrict__ out, _Float16* __restrict__ outh)
{
  __shared__ float s1[256], s2[256];
  int row = blockIdx.x, t = threadIdx.x;
  float v0 = a[(size_t)row*D_ + t]       + b[(size_t)row*D_ + t];
  float v1 = a[(size_t)row*D_ + 256 + t] + b[(size_t)row*D_ + 256 + t];
  s1[t] = v0 + v1; s2[t] = v0*v0 + v1*v1;
  __syncthreads();
  for (int st = 128; st > 0; st >>= 1) {
    if (t < st) { s1[t] += s1[t+st]; s2[t] += s2[t+st]; }
    __syncthreads();
  }
  float mean = s1[0] / D_;
  float var  = s2[0] / D_ - mean*mean;
  float r    = rsqrtf(var + 1e-5f);
  float o0 = (v0 - mean)*r*g[t]       + be[t];
  float o1 = (v1 - mean)*r*g[256 + t] + be[256 + t];
  out[(size_t)row*D_ + t]       = o0;
  out[(size_t)row*D_ + 256 + t] = o1;
  if (outh) {
    outh[(size_t)row*D_ + t]       = (_Float16)o0;
    outh[(size_t)row*D_ + 256 + t] = (_Float16)o1;
  }
}

// ---------------------------------------------------------------- router
__global__ void __launch_bounds__(256) gate_kernel(
    const float* __restrict__ x, const float* __restrict__ Wg,
    int* __restrict__ topi, float* __restrict__ topw)
{
  int wv = threadIdx.x >> 5, lane = threadIdx.x & 31;
  int n = blockIdx.x*8 + wv;
  float acc[E_];
#pragma unroll
  for (int e = 0; e < E_; ++e) acc[e] = 0.f;
  for (int d = lane; d < D_; d += 32) {
    float xv = x[(size_t)n*D_ + d];
    const float* wr = &Wg[(size_t)d*E_];
#pragma unroll
    for (int e = 0; e < E_; ++e) acc[e] += xv * wr[e];
  }
#pragma unroll
  for (int e = 0; e < E_; ++e)
#pragma unroll
    for (int m = 1; m < 32; m <<= 1) acc[e] += __shfl_xor(acc[e], m, 32);
  if (lane == 0) {
    float mx = acc[0];
#pragma unroll
    for (int e = 1; e < E_; ++e) mx = fmaxf(mx, acc[e]);
    float p[E_], s = 0.f;
#pragma unroll
    for (int e = 0; e < E_; ++e) { p[e] = __expf(acc[e] - mx); s += p[e]; }
#pragma unroll
    for (int e = 0; e < E_; ++e) p[e] /= s;
    int i0 = 0;
#pragma unroll
    for (int e = 1; e < E_; ++e) if (p[e] > p[i0]) i0 = e;
    int i1 = (i0 == 0) ? 1 : 0;
#pragma unroll
    for (int e = 0; e < E_; ++e) if (e != i0 && p[e] > p[i1]) i1 = e;
    float tt = p[i0] + p[i1];
    topi[n*2]   = i0; topw[n*2]   = p[i0] / tt;
    topi[n*2+1] = i1; topw[n*2+1] = p[i1] / tt;
  }
}

__global__ void scatter_kernel(const int* __restrict__ topi, const float* __restrict__ topw,
                               int* __restrict__ cnt, int* __restrict__ erows,
                               float* __restrict__ ewt)
{
  int n = blockIdx.x*256 + threadIdx.x;
  if (n >= NTOK) return;
#pragma unroll
  for (int j = 0; j < 2; ++j) {
    int e = topi[n*2 + j];
    int slot = atomicAdd(&cnt[e], 1);
    erows[(size_t)e*NTOK + slot] = n;
    ewt[(size_t)e*NTOK + slot]   = topw[n*2 + j];
  }
}

// ---------------------------------------------------------------- host
extern "C" void kernel_launch(void* const* d_in, const int* in_sizes, int n_in,
                              void* d_out, int out_size, void* d_ws, size_t ws_size,
                              hipStream_t stream) {
  (void)in_sizes; (void)n_in; (void)out_size; (void)ws_size;
  const float* src   = (const float*)d_in[0];
  const float* cosb  = (const float*)d_in[1];
  const float* sinb  = (const float*)d_in[2];
  const float* Wq    = (const float*)d_in[3];  const float* bq = (const float*)d_in[4];
  const float* Wk    = (const float*)d_in[5];  const float* bk = (const float*)d_in[6];
  const float* Wv    = (const float*)d_in[7];  const float* bv = (const float*)d_in[8];
  const float* Wo    = (const float*)d_in[9];  const float* bo = (const float*)d_in[10];
  const float* ln1_g = (const float*)d_in[11]; const float* ln1_b = (const float*)d_in[12];
  const float* ln2_g = (const float*)d_in[13]; const float* ln2_b = (const float*)d_in[14];
  const float* Wg    = (const float*)d_in[15];
  const float* W1    = (const float*)d_in[16]; const float* B1 = (const float*)d_in[17];
  const float* W2    = (const float*)d_in[18]; const float* B2 = (const float*)d_in[19];
  float* out = (float*)d_out;

  char* p = (char*)d_ws;
  auto alloc = [&](size_t bytes) -> char* {
    char* r = p; p += (bytes + 255) & ~(size_t)255; return r;
  };
  _Float16* srch = (_Float16*)alloc((size_t)NTOK*D_*2);
  _Float16* wqt  = (_Float16*)alloc((size_t)D_*D_*2);   // transposed f16 weights
  _Float16* wkt  = (_Float16*)alloc((size_t)D_*D_*2);
  _Float16* wvt  = (_Float16*)alloc((size_t)D_*D_*2);
  _Float16* wot  = (_Float16*)alloc((size_t)D_*D_*2);
  _Float16* w1t  = (_Float16*)alloc((size_t)E_*D_*F_*2); // per-expert [F][D]
  _Float16* w2t  = (_Float16*)alloc((size_t)E_*F_*D_*2); // per-expert [D][F]
  float*    qf   = (float*)   alloc((size_t)NTOK*D_*4);
  float*    kf   = (float*)   alloc((size_t)NTOK*D_*4);
  float*    vf   = (float*)   alloc((size_t)NTOK*D_*4);
  _Float16* qhb  = (_Float16*)alloc((size_t)NTOK*D_*2);
  _Float16* khb  = (_Float16*)alloc((size_t)NTOK*D_*2);
  _Float16* vtb  = (_Float16*)alloc((size_t)NTOK*D_*2);
  _Float16* ctxh = (_Float16*)alloc((size_t)NTOK*D_*2);
  _Float16* xh   = (_Float16*)alloc((size_t)NTOK*D_*2);
  _Float16* hh   = (_Float16*)alloc((size_t)NTOK*F_*2);
  int*      topi = (int*)     alloc((size_t)NTOK*2*4);
  float*    topw = (float*)   alloc((size_t)NTOK*2*4);
  int*      cnt  = (int*)     alloc((size_t)E_*4);
  int*      erows= (int*)     alloc((size_t)E_*NTOK*4);
  float*    ewt  = (float*)   alloc((size_t)E_*NTOK*4);
  // aliases: fp32 qkv buffers are dead after rope_reorder
  float* attn = qf;      // attention output projection result
  float* x    = kf;      // post-LN1 activations
  float* moe  = vf;      // MoE accumulator

  cvt_kernel<<<(NTOK*D_ + 255)/256, 256, 0, stream>>>(src, srch, NTOK*D_);
  // weight convert + transpose (once per launch; reused across all M tiles)
  cvt_t_kernel<<<(D_*D_ + 255)/256, 256, 0, stream>>>(Wq, wqt, D_, D_, D_*D_);
  cvt_t_kernel<<<(D_*D_ + 255)/256, 256, 0, stream>>>(Wk, wkt, D_, D_, D_*D_);
  cvt_t_kernel<<<(D_*D_ + 255)/256, 256, 0, stream>>>(Wv, wvt, D_, D_, D_*D_);
  cvt_t_kernel<<<(D_*D_ + 255)/256, 256, 0, stream>>>(Wo, wot, D_, D_, D_*D_);
  cvt_t_kernel<<<(E_*D_*F_ + 255)/256, 256, 0, stream>>>(W1, w1t, D_, F_, E_*D_*F_);
  cvt_t_kernel<<<(E_*F_*D_ + 255)/256, 256, 0, stream>>>(W2, w2t, F_, D_, E_*F_*D_);

  dim3 blk(128);
  dim3 gdd(D_/BN, NTOK/BM);   // 512-wide GEMMs over 4096 rows
  gemm_f16<0><<<gdd, blk, 0, stream>>>(srch, wqt, bq, qf, nullptr,
      NTOK, D_, D_, D_, D_, D_, nullptr, nullptr, nullptr);
  gemm_f16<0><<<gdd, blk, 0, stream>>>(srch, wkt, bk, kf, nullptr,
      NTOK, D_, D_, D_, D_, D_, nullptr, nullptr, nullptr);
  gemm_f16<0><<<gdd, blk, 0, stream>>>(srch, wvt, bv, vf, nullptr,
      NTOK, D_, D_, D_, D_, D_, nullptr, nullptr, nullptr);

  rope_reorder_kernel<<<(NTOK*D_ + 255)/256, 256, 0, stream>>>(
      qf, kf, vf, cosb, sinb, qhb, khb, vtb);

  attn_kernel<<<dim3(S_/128, B_*H_), 256, 0, stream>>>(qhb, khb, vtb, ctxh);

  gemm_f16<0><<<gdd, blk, 0, stream>>>(ctxh, wot, bo, attn, nullptr,
      NTOK, D_, D_, D_, D_, D_, nullptr, nullptr, nullptr);

  add_ln_kernel<<<NTOK, 256, 0, stream>>>(src, attn, ln1_g, ln1_b, x, xh);

  gate_kernel<<<NTOK/8, 256, 0, stream>>>(x, Wg, topi, topw);
  zero_i32<<<1, 32, 0, stream>>>(cnt, E_);
  zero_f32<<<(NTOK*D_ + 255)/256, 256, 0, stream>>>(moe, NTOK*D_);
  scatter_kernel<<<NTOK/256, 256, 0, stream>>>(topi, topw, cnt, erows, ewt);

  dim3 gup(F_/BN, NTOK/BM);
  for (int e = 0; e < E_; ++e) {
    gemm_f16<1><<<gup, blk, 0, stream>>>(
        xh, w1t + (size_t)e*D_*F_, B1 + (size_t)e*F_, nullptr, hh,
        NTOK, F_, D_, D_, D_, F_, erows + (size_t)e*NTOK, nullptr, cnt + e);
    gemm_f16<2><<<gdd, blk, 0, stream>>>(
        hh, w2t + (size_t)e*F_*D_, B2 + (size_t)e*D_, moe, nullptr,
        NTOK, D_, F_, F_, F_, D_, erows + (size_t)e*NTOK, ewt + (size_t)e*NTOK, cnt + e);
  }

  add_ln_kernel<<<NTOK, 256, 0, stream>>>(x, moe, ln2_g, ln2_b, out, nullptr);
}
